// TreeLSTMCellv2_29841432773236
// MI455X (gfx1250) — matrix operand order repaired and checked by reading, not compiled, not run on hardware
//
#include <hip/hip_runtime.h>

// Problem constants (match reference)
#define NN     32768
#define KK     8
#define HH     256
#define TWOH   512            // 2H (GEMM K dimension)
#define OUTC   1024           // fused output columns: f(256) | i(256) | o(256) | u(256)
#define ROWS   16             // rows per block (one WMMA M tile)
#define HSTR   520            // LDS row stride for h_combined (bf16 elems), 16B-aligned rows
#define CSTR   260            // LDS row stride for c_sum (f32 elems), 16B-aligned rows

typedef __attribute__((ext_vector_type(16))) __bf16 v16bf;
typedef __attribute__((ext_vector_type(8)))  float  v8f;

union Frag {
    v16bf bf;
    uint4 u4[2];
};

static __device__ __forceinline__ unsigned int pack2bf(float lo, float hi) {
    union { float f; unsigned int u; } a, b;
    a.f = lo; b.f = hi;
    unsigned int ru = (a.u + 0x7FFFu + ((a.u >> 16) & 1u)) >> 16;  // RNE
    unsigned int rv = (b.u + 0x7FFFu + ((b.u >> 16) & 1u)) >> 16;
    return (ru & 0xFFFFu) | (rv << 16);
}

static __device__ __forceinline__ float sigmoidf_fast(float x) {
    return 1.0f / (1.0f + __expf(-x));
}
static __device__ __forceinline__ float tanhf_fast(float x) {
    // 1 - 2/(e^{2x}+1); saturates correctly at +-1 for |x| large
    return 1.0f - 2.0f / (__expf(2.0f * x) + 1.0f);
}

// ---------------------------------------------------------------------------
// Kernel 1: pack [U_f_w ; U_iou_w] (f32, row-major [1024][512]) -> bf16 Wc.
// Wc[col][k]: col 0..255 = f (U_f_w rows), 256..511 = i, 512..767 = o,
// 768..1023 = u (U_iou_w rows). GEMM B operand is Wc read column-of-B = row.
// ---------------------------------------------------------------------------
__global__ __launch_bounds__(256) void pack_weights_bf16(
        const float* __restrict__ U_iou_w,   // [768][512]
        const float* __restrict__ U_f_w,     // [256][512]
        unsigned short* __restrict__ Wc) {   // [1024][512] bf16
    int i4 = blockIdx.x * 256 + threadIdx.x;     // float4 index, 0..131071
    int e   = i4 << 2;                           // element index
    int col = e >> 9;                            // 0..1023
    int k   = e & 511;
    const float* src = (col < 256) ? (U_f_w + (size_t)col * 512 + k)
                                   : (U_iou_w + (size_t)(col - 256) * 512 + k);
    float4 v = *(const float4*)src;
    unsigned int lo = pack2bf(v.x, v.y);
    unsigned int hi = pack2bf(v.z, v.w);
    ((unsigned int*)Wc)[i4 * 2 + 0] = lo;
    ((unsigned int*)Wc)[i4 * 2 + 1] = hi;
}

// ---------------------------------------------------------------------------
// Kernel 2: fused TreeLSTM cell. One block = 16 rows.
//   Phase 1: K-reduction of mailbox_h (sum|max -> bf16 LDS) + mailbox_c (sum).
//   Phase 2: [16 x 512] x [512 x 1024] bf16 WMMA GEMM, weights from L2.
//   Phase 3: in-register gate math, store h|c.
// ---------------------------------------------------------------------------
__global__ __launch_bounds__(256) void treelstm_fused(
        const float* __restrict__ mh,        // [N][8][256]
        const float* __restrict__ mc,        // [N][8][256]
        const unsigned short* __restrict__ Wc, // [1024][512] bf16
        const float* __restrict__ b_iou,     // [768]
        const float* __restrict__ U_f_b,     // [256]
        float* __restrict__ out) {           // [N][512] (h | c)
    __shared__ unsigned short lds_h[ROWS * HSTR];  // h_combined, bf16: [16][512(+pad)]
    __shared__ float          lds_c[ROWS * CSTR];  // c_sum, f32:     [16][256(+pad)]

    const int t    = threadIdx.x;
    const int row0 = blockIdx.x * ROWS;

    // ---------------- Phase 1: streaming K-reduction (float4) ----------------
    {
        const int quad = t & 63;        // float4 slot along H (0..63)
        const int rp   = t >> 6;        // 0..3
        #pragma unroll
        for (int i = 0; i < 4; ++i) {
            const int row = (i << 2) + rp;                 // 0..15
            const long n  = (long)(row0 + row);
            const float4* ph = (const float4*)mh + n * (KK * 64) + quad;
            const float4* pc = (const float4*)mc + n * (KK * 64) + quad;
            float4 s  = ph[0];
            float4 mx = s;
            float4 cs = pc[0];
            #pragma unroll
            for (int k = 1; k < KK; ++k) {
                float4 v  = ph[k * 64];
                float4 vc = pc[k * 64];
                s.x += v.x;  s.y += v.y;  s.z += v.z;  s.w += v.w;
                mx.x = fmaxf(mx.x, v.x); mx.y = fmaxf(mx.y, v.y);
                mx.z = fmaxf(mx.z, v.z); mx.w = fmaxf(mx.w, v.w);
                cs.x += vc.x; cs.y += vc.y; cs.z += vc.z; cs.w += vc.w;
            }
            const int col4 = quad << 2;
            unsigned int* hw = (unsigned int*)&lds_h[row * HSTR + col4];
            hw[0] = pack2bf(s.x, s.y);
            hw[1] = pack2bf(s.z, s.w);
            unsigned int* mw = (unsigned int*)&lds_h[row * HSTR + HH + col4];
            mw[0] = pack2bf(mx.x, mx.y);
            mw[1] = pack2bf(mx.z, mx.w);
            *(float4*)&lds_c[row * CSTR + col4] = cs;
        }
    }
    __syncthreads();

    // ---------------- Phase 2: WMMA GEMM [16x512] x [512x1024] ----------------
    const int lane = t & 31;
    const int w    = t >> 5;         // wave 0..7, owns j in [w*32, w*32+32) of each gate
    const int lh   = lane >> 4;      // lane half
    const int nsub = lane & 15;

    v8f acc[8];
    #pragma unroll
    for (int nt = 0; nt < 8; ++nt)
        #pragma unroll
        for (int r = 0; r < 8; ++r) acc[nt][r] = 0.0f;

    // Precompute per-ntile B column (col = region*256 + w*32 + sub*16 + nsub)
    int bcol[8];
    #pragma unroll
    for (int nt = 0; nt < 8; ++nt)
        bcol[nt] = (nt >> 1) * 256 + w * 32 + (nt & 1) * 16 + nsub;

    for (int kt = 0; kt < TWOH / 32; ++kt) {
        // A fragment (16-bit A 16x32 layout): lane half -> K offset 0/8, then +16
        Frag a;
        {
            const int m  = nsub;
            const int ka = kt * 32 + lh * 8;
            const unsigned short* hp = &lds_h[m * HSTR + ka];
            a.u4[0] = *(const uint4*)(hp);
            a.u4[1] = *(const uint4*)(hp + 16);
        }
        #pragma unroll
        for (int nt = 0; nt < 8; ++nt) {
            // B fragment (32x16): lane half -> K offset 0/8, then +16, of column bcol
            Frag b;
            const unsigned short* wp = Wc + (size_t)bcol[nt] * TWOH + kt * 32 + lh * 8;
            b.u4[0] = *(const uint4*)(wp);
            b.u4[1] = *(const uint4*)(wp + 16);
            acc[nt] = __builtin_amdgcn_wmma_f32_16x16x32_bf16(
                false, a.bf, false, b.bf, (short)0, acc[nt], false, false);
        }
    }

    // ---------------- Phase 3: gates, fully in-register ----------------
    #pragma unroll
    for (int jt = 0; jt < 2; ++jt) {
        const int j  = w * 32 + jt * 16 + nsub;    // 0..255 within H
        const float bF = U_f_b[j];
        const float bI = b_iou[j];
        const float bO = b_iou[256 + j];
        const float bU = b_iou[512 + j];
        v8f aF = acc[0 + jt];
        v8f aI = acc[2 + jt];
        v8f aO = acc[4 + jt];
        v8f aU = acc[6 + jt];
        #pragma unroll
        for (int r = 0; r < 8; ++r) {
            const int row = r + lh * 8;
            const float f    = sigmoidf_fast(aF[r] + bF);
            const float cagg = f * lds_c[row * CSTR + j];
            const float iv   = sigmoidf_fast(aI[r] + bI);
            const float uv   = tanhf_fast(aU[r] + bU);
            const float c    = iv * uv + cagg;
            const float ov   = sigmoidf_fast(aO[r] + bO);
            const float h    = ov * tanhf_fast(c);
            const long  ob   = (long)(row0 + row) * TWOH;
            out[ob + j]       = h;
            out[ob + HH + j]  = c;
        }
    }
}

extern "C" void kernel_launch(void* const* d_in, const int* in_sizes, int n_in,
                              void* d_out, int out_size, void* d_ws, size_t ws_size,
                              hipStream_t stream) {
    const float* mh      = (const float*)d_in[0];  // mailbox_h [N,8,256]
    const float* mc      = (const float*)d_in[1];  // mailbox_c [N,8,256]
    const float* U_iou_w = (const float*)d_in[2];  // [768,512]
    const float* b_iou   = (const float*)d_in[3];  // [1,768]
    const float* U_f_w   = (const float*)d_in[4];  // [256,512]
    const float* U_f_b   = (const float*)d_in[5];  // [256]
    float* out           = (float*)d_out;          // [N,512]
    unsigned short* Wc   = (unsigned short*)d_ws;  // 1 MB bf16 weight copy

    // Pack weights to bf16 (524288 elems / 4 per thread / 256 threads = 512 blocks)
    pack_weights_bf16<<<512, 256, 0, stream>>>(U_iou_w, U_f_w, Wc);

    // Fused TreeLSTM: 32768 / 16 rows per block = 2048 blocks
    treelstm_fused<<<NN / ROWS, 256, 0, stream>>>(mh, mc, Wc, b_iou, U_f_b, out);
}
